// TacoVCNet_65489661330148
// MI455X (gfx1250) — compile-verified, matrix operand order checked by reading, not run on hardware
//
#include <hip/hip_runtime.h>

// ---------------------------------------------------------------------------
// TacoVC inference for MI455X (gfx1250).
// bf16 WMMA 16x16x32 for every matrix op.  All weights are converted ONCE per
// call to bf16 pre-swizzled into the WMMA B-fragment register layout
// ([ntile][kchunk][lane][16 bf16], 32B contiguous per lane).  Weights
// (~52MB bf16) stay resident in the 192MB L2; the sequential LSTM scans are
// L2-bound, so halving weight traffic and making every fragment load a
// coalesced b128 is the dominant optimization.  GEMM kernels are barrier-free
// (no LDS): B tiles are consumed by exactly one wave, A chunks are tiny and
// L1-resident.
// ---------------------------------------------------------------------------

typedef __attribute__((ext_vector_type(16))) __bf16 v16bf;
typedef __attribute__((ext_vector_type(8)))  float  v8f;

#define B_SZ    32
#define T_SZ    512
#define FEAT_I  768
#define D_ENC   512
#define D_LAT   1024
#define D_SPK   256
#define D_AR    256
#define D_H     1024
#define D_O     80
#define CHUNK_T 64

#define ACT_NONE 0
#define ACT_RELU 1

// RNE f32->bf16 (integer path; used in the one-time weight pack).
__device__ __forceinline__ __bf16 f2bf(float f) {
  unsigned u = __builtin_bit_cast(unsigned, f);
  u += 0x7FFFu + ((u >> 16) & 1u);
  unsigned short h = (unsigned short)(u >> 16);
  return __builtin_bit_cast(__bf16, h);
}
__device__ __forceinline__ float sigm(float x) { return 1.0f / (1.0f + __expf(-x)); }

// ISA 16-bit operand layout: element i of v16bf in lane l covers
// K = i + 8*(l>=16) + 8*(i>=8)  (VGPR p: K=2p,2p+1; +8 upper lanes; +16 VGPRs 4-7).
__device__ __forceinline__ int kmap(int i, int hi) { return i + hi * 8 + ((i >= 8) ? 8 : 0); }

// A fragment (16x32) straight from global f32: two runs of 8 consecutive
// floats per lane -> global_load_b128 x4 + cvt.
__device__ __forceinline__ v16bf frag_A_g(const float* __restrict__ A, size_t rowoff,
                                          int kk, int lane) {
  const float* p = A + rowoff + kk + ((lane >> 4) * 8);
  v16bf a;
#pragma unroll
  for (int i = 0; i < 8; ++i) a[i] = (__bf16)p[i];
#pragma unroll
  for (int i = 8; i < 16; ++i) a[i] = (__bf16)p[i + 8];
  return a;
}
// Same but with a zero-fill predicate (conv edge padding).
__device__ __forceinline__ v16bf frag_A_gz(const float* __restrict__ A, size_t rowoff,
                                           int kk, int lane, bool ok) {
  const float* p = A + rowoff + kk + ((lane >> 4) * 8);
  v16bf a;
#pragma unroll
  for (int i = 0; i < 8; ++i) a[i] = ok ? (__bf16)p[i] : (__bf16)0.0f;
#pragma unroll
  for (int i = 8; i < 16; ++i) a[i] = ok ? (__bf16)p[i + 8] : (__bf16)0.0f;
  return a;
}
// A fragment from an LDS-resident bf16 matrix (contiguous 8-elem runs ->
// ds_load_b128, verified in disasm).
__device__ __forceinline__ v16bf frag_A_lds(const __bf16* lds, int pitch, int mbase,
                                            int kk, int lane) {
  int m = mbase + (lane & 15), hi = lane >> 4;
  v16bf a;
#pragma unroll
  for (int i = 0; i < 16; ++i) a[i] = lds[m * pitch + kk + kmap(i, hi)];
  return a;
}
__device__ __forceinline__ v8f wmma_bf16(v16bf a, v16bf b, v8f c) {
  return __builtin_amdgcn_wmma_f32_16x16x32_bf16(false, a, false, b, (short)0, c, false, false);
}

// ---------------------------------------------------------------------------
// One-time weight pack: W[K,N] (f32, ldw==N) -> bf16 swizzled
// out[ntile][kchunk][lane][i], i.e. each lane's B fragment is 32B contiguous.
// ---------------------------------------------------------------------------
__global__ void pack_w_kernel(const float* __restrict__ W, int kchunks, int N,
                              __bf16* __restrict__ out)
{
  size_t idx = (size_t)blockIdx.x * 256 + threadIdx.x;
  size_t total = (size_t)kchunks * 32 * (size_t)N;
  if (idx >= total) return;
  int i     = (int)(idx & 15);
  int lane  = (int)((idx >> 4) & 31);
  int kc    = (int)((idx >> 9) % (size_t)kchunks);
  int ntile = (int)(idx / ((size_t)kchunks << 9));
  int k = kc * 32 + kmap(i, lane >> 4);
  int n = ntile * 16 + (lane & 15);
  out[idx] = f2bf(W[(size_t)k * N + n]);
}

// ---------------------------------------------------------------------------
// Generic dual-input WMMA GEMM, barrier-free.  Block = 256 thr (8 wave32),
// tile 32(M)x128(N); wave w owns column tile w.  B fragments are direct
// v16bf (2x b128) loads from packed weights.
// C = act(A0@W0 [+ A1@W1] [+ addsrc] [+ bias]).
// tb_swz: A0 row m -> row (m&31)*T + t0 + (m>>5) (time-major gather).
// kcT = total kchunks in the packed weight, kcO = starting kchunk (row offset).
// ---------------------------------------------------------------------------
__global__ __launch_bounds__(256)
void wmma_gemm_kernel(const float* __restrict__ A0, const __bf16* __restrict__ W0,
                      int K0, int lda0, int kcT0, int kcO0,
                      const float* __restrict__ A1, const __bf16* __restrict__ W1,
                      int K1, int lda1, int kcT1, int kcO1,
                      const float* __restrict__ addsrc, const float* __restrict__ bias,
                      float* __restrict__ C, int ldc, int tb_swz, int t0, int act)
{
  int tid = threadIdx.x, wave = tid >> 5, lane = tid & 31;
  int ntileG = blockIdx.x * 8 + wave;
  v8f acc0 = {}; v8f acc1 = {};
  for (int pass = 0; pass < 2; ++pass) {
    int Kp = pass ? K1 : K0;
    if (Kp <= 0) continue;
    const float*  A  = pass ? A1 : A0;
    const __bf16* Ww = pass ? W1 : W0;
    int lda = pass ? lda1 : lda0;
    int kcT = pass ? kcT1 : kcT0;
    int kcO = pass ? kcO1 : kcO0;
    int m0 = blockIdx.y * 32 + (lane & 15);
    int m1 = m0 + 16;
    int sw = (pass == 0) ? tb_swz : 0;
    size_t r0 = (size_t)(sw ? ((m0 & 31) * T_SZ + t0 + (m0 >> 5)) : m0) * (size_t)lda;
    size_t r1 = (size_t)(sw ? ((m1 & 31) * T_SZ + t0 + (m1 >> 5)) : m1) * (size_t)lda;
    const __bf16* wb = Ww + (((size_t)ntileG * kcT + kcO) * 32 + lane) * 16;
    int nkc = Kp >> 5;
    for (int kc = 0; kc < nkc; ++kc) {
      v16bf b = *(const v16bf*)(wb + (size_t)kc * 512);
      acc0 = wmma_bf16(frag_A_g(A, r0, kc * 32, lane), b, acc0);
      acc1 = wmma_bf16(frag_A_g(A, r1, kc * 32, lane), b, acc1);
    }
  }
  int n = ntileG * 16 + (lane & 15);
  float bv = bias ? bias[n] : 0.0f;
#pragma unroll
  for (int mt = 0; mt < 2; ++mt) {
    v8f acc = mt ? acc1 : acc0;
#pragma unroll
    for (int r = 0; r < 8; ++r) {
      int mg = blockIdx.y * 32 + mt * 16 + r + 8 * (lane >> 4);
      float v = acc[r] + bv;
      if (addsrc) v += addsrc[(size_t)mg * ldc + n];
      if (act == ACT_RELU) v = fmaxf(v, 0.0f);
      C[(size_t)mg * ldc + n] = v;
    }
  }
}

// ---------------------------------------------------------------------------
// Conv1d k=5 SAME + bias + BN(eval) + ReLU as an 80-chunk accumulated WMMA
// GEMM (conv_w [5,512,512] packs naturally as a [2560,512] weight).
// ---------------------------------------------------------------------------
__global__ __launch_bounds__(256)
void conv5_bn_relu_kernel(const float* __restrict__ X, const __bf16* __restrict__ Wsw,
                          const float* __restrict__ cb, const float* __restrict__ sc,
                          const float* __restrict__ sh, float* __restrict__ Y)
{
  int tid = threadIdx.x, wave = tid >> 5, lane = tid & 31;
  int ntileG = blockIdx.x * 8 + wave;
  v8f acc0 = {}; v8f acc1 = {};
  const __bf16* wb = Wsw + ((size_t)ntileG * 80 * 32 + lane) * 16;
  int m0 = blockIdx.y * 32 + (lane & 15);
  int m1 = m0 + 16;
  int b0 = m0 >> 9, tt0 = m0 & 511;
  int b1 = m1 >> 9, tt1 = m1 & 511;
  for (int kcAll = 0; kcAll < 80; ++kcAll) {
    int tap = kcAll >> 4;
    int kk  = (kcAll & 15) * 32;
    v16bf b = *(const v16bf*)(wb + (size_t)kcAll * 512);
    int ts0 = tt0 + tap - 2, ts1 = tt1 + tap - 2;
    bool ok0 = (ts0 >= 0) && (ts0 < T_SZ);
    bool ok1 = (ts1 >= 0) && (ts1 < T_SZ);
    size_t r0 = ((size_t)(b0 << 9) + (size_t)(ok0 ? ts0 : 0)) * D_ENC;
    size_t r1 = ((size_t)(b1 << 9) + (size_t)(ok1 ? ts1 : 0)) * D_ENC;
    acc0 = wmma_bf16(frag_A_gz(X, r0, kk, lane, ok0), b, acc0);
    acc1 = wmma_bf16(frag_A_gz(X, r1, kk, lane, ok1), b, acc1);
  }
  int n = ntileG * 16 + (lane & 15);
  float cbv = cb[n], scv = sc[n], shv = sh[n];
#pragma unroll
  for (int mt = 0; mt < 2; ++mt) {
    v8f acc = mt ? acc1 : acc0;
#pragma unroll
    for (int r = 0; r < 8; ++r) {
      int mg = blockIdx.y * 32 + mt * 16 + r + 8 * (lane >> 4);
      float v = (acc[r] + cbv) * scv + shv;
      Y[(size_t)mg * D_ENC + n] = fmaxf(v, 0.0f);
    }
  }
}

// Encoder LSTM cell (elementwise) + fused cond[t] = h + g write.
__global__ void enc_cell_kernel(const float* __restrict__ gates, const float* __restrict__ g,
                                float* __restrict__ h, float* __restrict__ c,
                                float* __restrict__ cond_t)
{
  int e = blockIdx.x * blockDim.x + threadIdx.x;
  if (e >= B_SZ * D_LAT) return;
  int b = e >> 10, j = e & (D_LAT - 1);
  const float* gr = gates + (size_t)b * 4 * D_LAT;
  float cc = sigm(gr[D_LAT + j]) * c[e] + sigm(gr[j]) * tanhf(gr[2 * D_LAT + j]);
  float hh = sigm(gr[3 * D_LAT + j]) * tanhf(cc);
  c[e] = cc; h[e] = hh;
  cond_t[e] = hh + g[e];
}

// ---------------------------------------------------------------------------
// Fused LSTMP cell + projection.  Every WG recomputes the cheap cell into LDS
// as bf16 (shared WMMA A operand -> ds_load_b128 fragments), then does its
// 128-column slice of z = tanh(h @ Wp + bp) with packed-Wp B fragments from
// global.  Only WG 0 commits h/c; c ping-pongs across steps.
// ---------------------------------------------------------------------------
#define HPITCH (D_H + 16)
__global__ __launch_bounds__(256)
void lstmp_cell_proj_kernel(const float* __restrict__ gates, const float* __restrict__ c_in,
                            const __bf16* __restrict__ Wp, const float* __restrict__ bp,
                            float* __restrict__ c_out, float* __restrict__ h_out,
                            float* __restrict__ z_out)
{
  __shared__ __bf16 Hs[B_SZ * HPITCH];      // 32x1024 bf16 (~66KB of 320KB/WGP)
  int tid = threadIdx.x, wave = tid >> 5, lane = tid & 31;
  for (int e = tid; e < B_SZ * D_H; e += 256) {
    int b = e >> 10, j = e & (D_H - 1);
    const float* gr = gates + (size_t)b * 4 * D_H;
    float cc = sigm(gr[D_H + j]) * c_in[e] + sigm(gr[j]) * tanhf(gr[2 * D_H + j]);
    float hh = sigm(gr[3 * D_H + j]) * tanhf(cc);
    Hs[b * HPITCH + j] = f2bf(hh);
    if (blockIdx.x == 0) { c_out[e] = cc; h_out[e] = hh; }
  }
  __syncthreads();
  int ntileG = blockIdx.x * 8 + wave;
  const __bf16* wb = Wp + ((size_t)ntileG * 32 * 32 + lane) * 16;   // kcT = 32
  v8f acc0 = {}; v8f acc1 = {};
  for (int kc = 0; kc < 32; ++kc) {
    v16bf b = *(const v16bf*)(wb + (size_t)kc * 512);
    acc0 = wmma_bf16(frag_A_lds(Hs, HPITCH, 0, kc * 32, lane), b, acc0);
    acc1 = wmma_bf16(frag_A_lds(Hs, HPITCH, 16, kc * 32, lane), b, acc1);
  }
  int n = ntileG * 16 + (lane & 15);
  float bv = bp[n];
#pragma unroll
  for (int mt = 0; mt < 2; ++mt) {
    v8f acc = mt ? acc1 : acc0;
#pragma unroll
    for (int r = 0; r < 8; ++r) {
      int m = mt * 16 + r + 8 * (lane >> 4);
      z_out[(size_t)m * D_H + n] = tanhf(acc[r] + bv);
    }
  }
}

// Prenet: a = relu(relu(prev@W1+b1)@W2+b2).  Tiny (32x80->256->256): one WG.
__global__ __launch_bounds__(256)
void prenet_kernel(const float* __restrict__ prev, const float* __restrict__ W1,
                   const float* __restrict__ b1, const float* __restrict__ W2,
                   const float* __restrict__ b2, float* __restrict__ a_out)
{
  __shared__ float P[B_SZ * D_O];
  __shared__ float A1s[B_SZ * D_AR];
  int tid = threadIdx.x;
  for (int e = tid; e < B_SZ * D_O; e += 256) P[e] = prev[e];
  __syncthreads();
  for (int e = tid; e < B_SZ * D_AR; e += 256) {
    int b = e >> 8, j = e & 255;
    float s = b1[j];
    for (int k = 0; k < D_O; ++k) s += P[b * D_O + k] * W1[k * D_AR + j];
    A1s[e] = fmaxf(s, 0.0f);
  }
  __syncthreads();
  for (int e = tid; e < B_SZ * D_AR; e += 256) {
    int b = e >> 8, j = e & 255;
    float s = b2[j];
    for (int k = 0; k < D_AR; ++k) s += A1s[b * D_AR + k] * W2[k * D_AR + j];
    a_out[e] = fmaxf(s, 0.0f);
  }
}

// Output projection (N=80, not WMMA-shaped) + AR feedback normalization.
__global__ __launch_bounds__(256)
void outproj_kernel(const float* __restrict__ z1, const float* __restrict__ Wo,
                    const float* __restrict__ bo, const float* __restrict__ mean,
                    const float* __restrict__ scale, float* __restrict__ out,
                    float* __restrict__ prev, int t)
{
  int tid = threadIdx.x;
  for (int e = tid; e < B_SZ * D_O; e += 256) {
    int b = e / D_O, n = e % D_O;
    float s = bo[n];
    const float* zr = z1 + (size_t)b * D_H;
    for (int k = 0; k < D_H; ++k) s += zr[k] * Wo[k * D_O + n];
    out[(size_t)b * T_SZ * D_O + (size_t)t * D_O + n] = s;
    prev[e] = (s - mean[n]) / scale[n];
  }
}

__global__ void zero_kernel(float* __restrict__ p, int n) {
  int i = blockIdx.x * 256 + threadIdx.x;
  if (i < n) p[i] = 0.0f;
}
__global__ void lens_kernel(const int* __restrict__ l, float* __restrict__ dst) {
  int i = threadIdx.x;
  if (i < B_SZ) dst[i] = (float)l[i] * 1.0f;   // RATIO == 1.0
}

// ---------------------------------------------------------------------------
extern "C" void kernel_launch(void* const* d_in, const int* in_sizes, int n_in,
                              void* d_out, int out_size, void* d_ws, size_t ws_size,
                              hipStream_t stream)
{
  (void)in_sizes; (void)n_in; (void)out_size; (void)ws_size;
  const float* features = (const float*)d_in[0];
  const int*   lens     = (const int*)  d_in[1];
  const float* spk      = (const float*)d_in[2];
  const float* enc_fc_w = (const float*)d_in[3];
  const float* enc_fc_b = (const float*)d_in[4];
  const float* conv_w[3] = {(const float*)d_in[5],  (const float*)d_in[9],  (const float*)d_in[13]};
  const float* conv_b[3] = {(const float*)d_in[6],  (const float*)d_in[10], (const float*)d_in[14]};
  const float* bn_s[3]   = {(const float*)d_in[7],  (const float*)d_in[11], (const float*)d_in[15]};
  const float* bn_h[3]   = {(const float*)d_in[8],  (const float*)d_in[12], (const float*)d_in[16]};
  const float* eWx = (const float*)d_in[17];
  const float* eWh = (const float*)d_in[18];
  const float* eb  = (const float*)d_in[19];
  const float* gW  = (const float*)d_in[20];
  const float* gb  = (const float*)d_in[21];
  const float* pw1 = (const float*)d_in[22];
  const float* pb1 = (const float*)d_in[23];
  const float* pw2 = (const float*)d_in[24];
  const float* pb2 = (const float*)d_in[25];
  const float* dWx[2] = {(const float*)d_in[26], (const float*)d_in[31]};
  const float* dWh[2] = {(const float*)d_in[27], (const float*)d_in[32]};
  const float* db_[2] = {(const float*)d_in[28], (const float*)d_in[33]};
  const float* dWp[2] = {(const float*)d_in[29], (const float*)d_in[34]};
  const float* dbp[2] = {(const float*)d_in[30], (const float*)d_in[35]};
  const float* projw  = (const float*)d_in[36];
  const float* projb  = (const float*)d_in[37];
  const float* meanp  = (const float*)d_in[38];
  const float* scalep = (const float*)d_in[39];
  float* out = (float*)d_out;

  // ---- workspace carve (~271MB) ----
  float* ws = (float*)d_ws;
  size_t off = 0;
  auto carve  = [&](size_t n) -> float*  { float* p = ws + off; off += (n + 63) & ~(size_t)63; return p; };
  auto carveB = [&](size_t n) -> __bf16* { return (__bf16*)carve((n + 1) / 2); };
  float* buf0   = carve((size_t)B_SZ * T_SZ * D_ENC);
  float* buf1   = carve((size_t)B_SZ * T_SZ * D_ENC);
  float* cond   = carve((size_t)T_SZ * B_SZ * D_LAT);       // [T,B,1024] time-major
  float* gvec   = carve((size_t)B_SZ * D_LAT);
  float* gX     = carve((size_t)CHUNK_T * B_SZ * 4 * D_LAT);
  float* cpre   = carve((size_t)CHUNK_T * B_SZ * 4 * D_H);
  float* gates0 = carve((size_t)B_SZ * 4 * D_H);
  float* gates1 = carve((size_t)B_SZ * 4 * D_H);
  float* z0     = carve((size_t)B_SZ * D_H);
  float* z1     = carve((size_t)B_SZ * D_H);
  // packed bf16 weights
  __bf16* eFCp  = carveB((size_t)FEAT_I * D_ENC);
  __bf16* convp[3] = { carveB((size_t)5 * D_ENC * D_ENC),
                       carveB((size_t)5 * D_ENC * D_ENC),
                       carveB((size_t)5 * D_ENC * D_ENC) };
  __bf16* eWxp  = carveB((size_t)D_ENC * 4 * D_LAT);
  __bf16* eWhp  = carveB((size_t)D_LAT * 4 * D_LAT);
  __bf16* gWp   = carveB((size_t)D_SPK * D_LAT);
  __bf16* dWx0p = carveB((size_t)(D_LAT + D_AR) * 4 * D_H);
  __bf16* dWh0p = carveB((size_t)D_H * 4 * D_H);
  __bf16* dWx1p = carveB((size_t)D_H * 4 * D_H);
  __bf16* dWh1p = carveB((size_t)D_H * 4 * D_H);
  __bf16* dWp0p = carveB((size_t)D_H * D_H);
  __bf16* dWp1p = carveB((size_t)D_H * D_H);
  // recurrent state (zeroed every call)
  size_t stateBegin = off;
  float* hE  = carve((size_t)B_SZ * D_LAT);
  float* cE  = carve((size_t)B_SZ * D_LAT);
  float* h0  = carve((size_t)B_SZ * D_H);
  float* c0a = carve((size_t)B_SZ * D_H);
  float* c0b = carve((size_t)B_SZ * D_H);
  float* h1  = carve((size_t)B_SZ * D_H);
  float* c1a = carve((size_t)B_SZ * D_H);
  float* c1b = carve((size_t)B_SZ * D_H);
  float* aP  = carve((size_t)B_SZ * D_AR);
  float* prev = carve((size_t)B_SZ * D_O);
  size_t stateEnd = off;

  int nstate = (int)(stateEnd - stateBegin);
  zero_kernel<<<dim3((nstate + 255) / 256), 256, 0, stream>>>(ws + stateBegin, nstate);
  lens_kernel<<<1, 32, 0, stream>>>(lens, out + (size_t)B_SZ * T_SZ * D_O);

  // ---- one-time weight pack to swizzled bf16 ----
  auto pack = [&](const float* W, int K, int N, __bf16* dst) {
    size_t total = (size_t)K * N;
    pack_w_kernel<<<dim3((unsigned)((total + 255) / 256)), 256, 0, stream>>>(W, K / 32, N, dst);
  };
  pack(enc_fc_w, FEAT_I, D_ENC, eFCp);
  for (int i = 0; i < 3; ++i) pack(conv_w[i], 5 * D_ENC, D_ENC, convp[i]);
  pack(eWx, D_ENC, 4 * D_LAT, eWxp);
  pack(eWh, D_LAT, 4 * D_LAT, eWhp);
  pack(gW, D_SPK, D_LAT, gWp);
  pack(dWx[0], D_LAT + D_AR, 4 * D_H, dWx0p);
  pack(dWh[0], D_H, 4 * D_H, dWh0p);
  pack(dWx[1], D_H, 4 * D_H, dWx1p);
  pack(dWh[1], D_H, 4 * D_H, dWh1p);
  pack(dWp[0], D_H, D_H, dWp0p);
  pack(dWp[1], D_H, D_H, dWp1p);

  // ---- encoder: segFC ----
  wmma_gemm_kernel<<<dim3(D_ENC / 128, (B_SZ * T_SZ) / 32), 256, 0, stream>>>(
      features, eFCp, FEAT_I, FEAT_I, FEAT_I / 32, 0,
      nullptr, nullptr, 0, 0, 0, 0,
      nullptr, enc_fc_b, buf0, D_ENC, 0, 0, ACT_NONE);

  // ---- 3x (conv k5 + BN + ReLU) ----
  const float* cin = buf0; float* cob = buf1;
  for (int i = 0; i < 3; ++i) {
    conv5_bn_relu_kernel<<<dim3(D_ENC / 128, (B_SZ * T_SZ) / 32), 256, 0, stream>>>(
        cin, convp[i], conv_b[i], bn_s[i], bn_h[i], cob);
    float* tmp = cob; cob = (float*)cin; cin = tmp;
  }
  const float* convOut = cin;   // buf1 after 3 hops

  // ---- global conditioning g = spk@gW + gb ----
  wmma_gemm_kernel<<<dim3(D_LAT / 128, 1), 256, 0, stream>>>(
      spk, gWp, D_SPK, D_SPK, D_SPK / 32, 0,
      nullptr, nullptr, 0, 0, 0, 0,
      nullptr, gb, gvec, D_LAT, 0, 0, ACT_NONE);

  // ---- encoder LSTM scan: x@Wx precomputed per 64-step chunk ----
  for (int ch = 0; ch < T_SZ / CHUNK_T; ++ch) {
    int t0 = ch * CHUNK_T;
    wmma_gemm_kernel<<<dim3((4 * D_LAT) / 128, (CHUNK_T * B_SZ) / 32), 256, 0, stream>>>(
        convOut, eWxp, D_ENC, D_ENC, D_ENC / 32, 0,
        nullptr, nullptr, 0, 0, 0, 0,
        nullptr, eb, gX, 4 * D_LAT, /*tb_swz=*/1, t0, ACT_NONE);
    for (int s = 0; s < CHUNK_T; ++s) {
      int t = t0 + s;
      wmma_gemm_kernel<<<dim3((4 * D_LAT) / 128, 1), 256, 0, stream>>>(
          hE, eWhp, D_LAT, D_LAT, D_LAT / 32, 0,
          nullptr, nullptr, 0, 0, 0, 0,
          gX + (size_t)s * B_SZ * 4 * D_LAT, nullptr, gates0, 4 * D_LAT, 0, 0, ACT_NONE);
      enc_cell_kernel<<<dim3((B_SZ * D_LAT + 255) / 256), 256, 0, stream>>>(
          gates0, gvec, hE, cE, cond + (size_t)t * B_SZ * D_LAT);
    }
  }

  // ---- autoregressive decoder: cond@Wx0 precomputed per chunk, 6 kernels/step ----
  for (int ch = 0; ch < T_SZ / CHUNK_T; ++ch) {
    int t0 = ch * CHUNK_T;
    wmma_gemm_kernel<<<dim3((4 * D_H) / 128, (CHUNK_T * B_SZ) / 32), 256, 0, stream>>>(
        cond + (size_t)t0 * B_SZ * D_LAT, dWx0p, D_LAT, D_LAT, (D_LAT + D_AR) / 32, 0,
        nullptr, nullptr, 0, 0, 0, 0,
        nullptr, db_[0], cpre, 4 * D_H, 0, 0, ACT_NONE);
    for (int s = 0; s < CHUNK_T; ++s) {
      int t = t0 + s;
      prenet_kernel<<<1, 256, 0, stream>>>(prev, pw1, pb1, pw2, pb2, aP);
      // gates0 = a@Wx0[1024:1280] + h0@Wh0 + cpre[s]   (cpre carries b0)
      wmma_gemm_kernel<<<dim3((4 * D_H) / 128, 1), 256, 0, stream>>>(
          aP, dWx0p, D_AR, D_AR, (D_LAT + D_AR) / 32, D_LAT / 32,
          h0, dWh0p, D_H, D_H, D_H / 32, 0,
          cpre + (size_t)s * B_SZ * 4 * D_H, nullptr, gates0, 4 * D_H, 0, 0, ACT_NONE);
      const float* c0i = (t & 1) ? c0b : c0a; float* c0o = (t & 1) ? c0a : c0b;
      lstmp_cell_proj_kernel<<<dim3(D_H / 128), 256, 0, stream>>>(
          gates0, c0i, dWp0p, dbp[0], c0o, h0, z0);
      wmma_gemm_kernel<<<dim3((4 * D_H) / 128, 1), 256, 0, stream>>>(
          z0, dWx1p, D_H, D_H, D_H / 32, 0,
          h1, dWh1p, D_H, D_H, D_H / 32, 0,
          nullptr, db_[1], gates1, 4 * D_H, 0, 0, ACT_NONE);
      const float* c1i = (t & 1) ? c1b : c1a; float* c1o = (t & 1) ? c1a : c1b;
      lstmp_cell_proj_kernel<<<dim3(D_H / 128), 256, 0, stream>>>(
          gates1, c1i, dWp1p, dbp[1], c1o, h1, z1);
      outproj_kernel<<<1, 256, 0, stream>>>(z1, projw, projb, meanp, scalep, out, prev, t);
    }
  }
}
